// Attention_55628416418357
// MI455X (gfx1250) — matrix operand order repaired and checked by reading, not compiled, not run on hardware
//
#include <hip/hip_runtime.h>

typedef __attribute__((ext_vector_type(16))) __bf16 v16bf;
typedef __attribute__((ext_vector_type(8)))  __bf16 v8bf;
typedef __attribute__((ext_vector_type(8)))  float  v8f;

#define DIM    768
#define NHEADS 12
#define HDIM   64
#define BATCH  4
#define SEQ    2048
#define TOKENS (BATCH*SEQ)     // 8192
#define QKVOUT (3*DIM)         // 2304
#define ATTN_SCALE 0.125f      // 64^-0.5

union F8 { v8f v; float f[8]; };

__device__ __forceinline__ unsigned short bf16_of(float x) {
  union { float f; unsigned u; } c; c.f = x;
  unsigned u = c.u + 0x7FFFu + ((c.u >> 16) & 1u);   // round-to-nearest-even
  return (unsigned short)(u >> 16);
}

__device__ __forceinline__ v16bf cat8(v8bf lo, v8bf hi) {
  return __builtin_shufflevector(lo, hi, 0,1,2,3,4,5,6,7,8,9,10,11,12,13,14,15);
}

__device__ __forceinline__ v8f wmma_bf16(v16bf a, v16bf b, v8f c) {
  // (neg_a, A, neg_b, B, c_mod, C, reuse_a, reuse_b)
  return __builtin_amdgcn_wmma_f32_16x16x32_bf16(false, a, false, b, (short)0, c, false, false);
}

// ---------------------------------------------------------------- fp32 -> bf16
__global__ void cvt_bf16_kernel(const float* __restrict__ in,
                                unsigned short* __restrict__ out, int n) {
  int i = blockIdx.x * blockDim.x + threadIdx.x;
  if (i < n) out[i] = bf16_of(in[i]);
}

// ---------------------------------------------------------------- QKV GEMM
// out[t][o] = sum_i x[t][i] * w_qkv[o][i]; writes Q,K as [B,H,N,64], V as [B,H,64,N]
__global__ void __launch_bounds__(128)
qkv_gemm_kernel(const unsigned short* __restrict__ xb_u,
                const unsigned short* __restrict__ wb_u,
                unsigned short* __restrict__ Qb,
                unsigned short* __restrict__ Kb,
                unsigned short* __restrict__ Vtb) {
  const __bf16* xb = (const __bf16*)xb_u;
  const __bf16* wb = (const __bf16*)wb_u;
  const int lane = threadIdx.x & 31;
  const int wv   = threadIdx.x >> 5;     // 4 waves / block
  const int hf   = lane >> 4;
  const int l16  = lane & 15;
  const int rowbase = blockIdx.y * 16;              // token tile
  const int colbase = blockIdx.x * 256 + wv * 64;   // output-feature tile

  F8 acc[4];
  #pragma unroll
  for (int t = 0; t < 4; ++t)
    #pragma unroll
    for (int r = 0; r < 8; ++r) acc[t].f[r] = 0.0f;

  const __bf16* arow = xb + (rowbase + l16) * DIM;
  for (int k0 = 0; k0 < DIM; k0 += 32) {
    v16bf A = *(const v16bf*)(arow + k0 + 16*hf);
    #pragma unroll
    for (int t = 0; t < 4; ++t) {
      v16bf Bop = *(const v16bf*)(wb + (colbase + t*16 + l16) * DIM + k0 + 16*hf);
      acc[t].v = wmma_bf16(A, Bop, acc[t].v);
    }
  }

  #pragma unroll
  for (int t = 0; t < 4; ++t) {
    const int o0  = colbase + t*16;
    const int sec = o0 / DIM;                 // 0=Q 1=K 2=V (tile never crosses)
    const int rem = o0 - sec * DIM;
    const int hh  = rem / HDIM;
    const int d   = (rem & (HDIM - 1)) + l16;
    #pragma unroll
    for (int r = 0; r < 8; ++r) {
      const int tok = rowbase + r + 8*hf;
      const int bb  = tok >> 11;
      const int nn  = tok & (SEQ - 1);
      unsigned short val = bf16_of(acc[t].f[r]);
      if (sec == 0)      Qb [((bb*NHEADS + hh)*SEQ + nn)*HDIM + d] = val;
      else if (sec == 1) Kb [((bb*NHEADS + hh)*SEQ + nn)*HDIM + d] = val;
      else               Vtb[((bb*NHEADS + hh)*HDIM + d)*SEQ + nn] = val;
    }
  }
}

// ---------------------------------------------------------------- flash attention
// One wave per 16 query rows; online softmax over key chunks of 32.
__global__ void __launch_bounds__(128)
flash_attn_kernel(const unsigned short* __restrict__ Qu,
                  const unsigned short* __restrict__ Ku,
                  const unsigned short* __restrict__ Vu,
                  unsigned short* __restrict__ Ob) {
  __shared__ __align__(64) unsigned short sP[4][16*32];  // per-wave P staging
  const __bf16* Qb = (const __bf16*)Qu;
  const __bf16* Kb = (const __bf16*)Ku;
  const __bf16* Vb = (const __bf16*)Vu;
  const int lane = threadIdx.x & 31;
  const int wv   = threadIdx.x >> 5;
  const int hf   = lane >> 4;
  const int l16  = lane & 15;
  const int bh   = blockIdx.y;                    // b*12+h
  const int qrow = (blockIdx.x * 4 + wv) * 16;    // query-row base

  const __bf16* Qh = Qb + (size_t)bh * SEQ * HDIM;
  const __bf16* Kh = Kb + (size_t)bh * SEQ * HDIM;
  const __bf16* Vh = Vb + (size_t)bh * HDIM * SEQ;

  // Q A-operands (dim chunks 0..31 and 32..63), resident all loop long
  const __bf16* qp = Qh + (qrow + l16) * HDIM;
  v16bf Aq0 = *(const v16bf*)(qp + 16*hf);
  v16bf Aq1 = *(const v16bf*)(qp + 32 + 16*hf);

  F8 O[4]; float mrow[8], lrow[8];
  #pragma unroll
  for (int g = 0; g < 4; ++g)
    #pragma unroll
    for (int r = 0; r < 8; ++r) O[g].f[r] = 0.0f;
  #pragma unroll
  for (int r = 0; r < 8; ++r) { mrow[r] = -1e30f; lrow[r] = 0.0f; }

  unsigned short* myP = &sP[wv][0];

  for (int kc = 0; kc < SEQ; kc += 32) {
    // ---- S = (Q K^T) * scale, two 16x16 tiles (key groups 0,1)
    F8 S0, S1;
    #pragma unroll
    for (int r = 0; r < 8; ++r) { S0.f[r] = 0.0f; S1.f[r] = 0.0f; }
    {
      const __bf16* kr0 = Kh + (kc + l16) * HDIM;        // key group 0
      S0.v = wmma_bf16(Aq0, *(const v16bf*)(kr0 + 16*hf),       S0.v);
      S0.v = wmma_bf16(Aq1, *(const v16bf*)(kr0 + 32 + 16*hf),  S0.v);
      const __bf16* kr1 = Kh + (kc + 16 + l16) * HDIM;   // key group 1
      S1.v = wmma_bf16(Aq0, *(const v16bf*)(kr1 + 16*hf),       S1.v);
      S1.v = wmma_bf16(Aq1, *(const v16bf*)(kr1 + 32 + 16*hf),  S1.v);
    }

    // ---- online softmax (row = r + 8*hf, column = lane&15 per tile)
    float alpha[8];
    #pragma unroll
    for (int r = 0; r < 8; ++r) {
      float s0 = S0.f[r] * ATTN_SCALE;
      float s1 = S1.f[r] * ATTN_SCALE;
      float t  = fmaxf(s0, s1);
      t = fmaxf(t, __shfl_xor(t, 1, 32));
      t = fmaxf(t, __shfl_xor(t, 2, 32));
      t = fmaxf(t, __shfl_xor(t, 4, 32));
      t = fmaxf(t, __shfl_xor(t, 8, 32));
      float mnew = fmaxf(mrow[r], t);
      alpha[r] = __expf(mrow[r] - mnew);
      mrow[r]  = mnew;
      float p0 = __expf(s0 - mnew);
      float p1 = __expf(s1 - mnew);
      S0.f[r] = p0; S1.f[r] = p1;
      float rs = p0 + p1;
      rs += __shfl_xor(rs, 1, 32);
      rs += __shfl_xor(rs, 2, 32);
      rs += __shfl_xor(rs, 4, 32);
      rs += __shfl_xor(rs, 8, 32);
      lrow[r] = alpha[r] * lrow[r] + rs;
    }
    #pragma unroll
    for (int g = 0; g < 4; ++g)
      #pragma unroll
      for (int r = 0; r < 8; ++r) O[g].f[r] *= alpha[r];

    // ---- P: C-layout f32 -> A-layout bf16 via wave-private LDS slice
    #pragma unroll
    for (int r = 0; r < 8; ++r) {
      const int m = r + 8*hf;
      myP[m*32 + l16]      = bf16_of(S0.f[r]);
      myP[m*32 + 16 + l16] = bf16_of(S1.f[r]);
    }
    asm volatile("s_wait_dscnt 0" ::: "memory");
    const __bf16* pp = (const __bf16*)myP + l16*32;
    v16bf Ap = *(const v16bf*)(pp + 16*hf);

    // ---- O += P * V  (V stored transposed [64][SEQ] -> contiguous B operand)
    #pragma unroll
    for (int g = 0; g < 4; ++g) {
      const __bf16* vp = Vh + (g*16 + l16) * SEQ + kc + 16*hf;
      O[g].v = wmma_bf16(Ap, *(const v16bf*)vp, O[g].v);
    }
  }

  // ---- normalize and store to attn buffer [B, N, 768] (bf16)
  const int b = bh / NHEADS, h = bh - b*NHEADS;
  #pragma unroll
  for (int g = 0; g < 4; ++g)
    #pragma unroll
    for (int r = 0; r < 8; ++r) {
      const int m = qrow + r + 8*hf;
      Ob[(size_t)(b*SEQ + m)*DIM + h*HDIM + g*16 + l16] = bf16_of(O[g].f[r] / lrow[r]);
    }
}

// ---------------------------------------------------------------- output projection
__global__ void __launch_bounds__(128)
proj_gemm_kernel(const unsigned short* __restrict__ au,
                 const unsigned short* __restrict__ wu,
                 const float* __restrict__ bias,
                 float* __restrict__ out) {
  const __bf16* ab = (const __bf16*)au;
  const __bf16* wb = (const __bf16*)wu;
  const int lane = threadIdx.x & 31;
  const int wv   = threadIdx.x >> 5;
  const int hf   = lane >> 4;
  const int l16  = lane & 15;
  const int rowbase = blockIdx.y * 16;
  const int colbase = blockIdx.x * 256 + wv * 64;

  F8 acc[4];
  #pragma unroll
  for (int t = 0; t < 4; ++t)
    #pragma unroll
    for (int r = 0; r < 8; ++r) acc[t].f[r] = 0.0f;

  const __bf16* arow = ab + (rowbase + l16) * DIM;
  for (int k0 = 0; k0 < DIM; k0 += 32) {
    v16bf A = *(const v16bf*)(arow + k0 + 16*hf);
    #pragma unroll
    for (int t = 0; t < 4; ++t) {
      v16bf Bop = *(const v16bf*)(wb + (colbase + t*16 + l16) * DIM + k0 + 16*hf);
      acc[t].v = wmma_bf16(A, Bop, acc[t].v);
    }
  }

  #pragma unroll
  for (int t = 0; t < 4; ++t) {
    const int o = colbase + t*16 + l16;
    const float bv = bias[o];
    #pragma unroll
    for (int r = 0; r < 8; ++r) {
      const int tok = rowbase + r + 8*hf;
      out[(size_t)tok*DIM + o] = acc[t].f[r] + bv;
    }
  }
}

// ---------------------------------------------------------------- launch
extern "C" void kernel_launch(void* const* d_in, const int* in_sizes, int n_in,
                              void* d_out, int out_size, void* d_ws, size_t ws_size,
                              hipStream_t stream) {
  const float* x      = (const float*)d_in[0];
  const float* w_qkv  = (const float*)d_in[1];
  const float* w_proj = (const float*)d_in[2];
  const float* b_proj = (const float*)d_in[3];
  float* out = (float*)d_out;

  char* ws = (char*)d_ws;
  const size_t XB = (size_t)TOKENS * DIM * 2;     // x bf16
  const size_t WQ = (size_t)QKVOUT * DIM * 2;     // w_qkv bf16
  const size_t WP = (size_t)DIM * DIM * 2;        // w_proj bf16
  const size_t QB = (size_t)BATCH * NHEADS * SEQ * HDIM * 2;  // one of Q/K/Vt
  unsigned short* xb  = (unsigned short*)(ws);
  unsigned short* wqb = (unsigned short*)(ws + XB);
  unsigned short* wpb = (unsigned short*)(ws + XB + WQ);
  unsigned short* Qb  = (unsigned short*)(ws + XB + WQ + WP);
  unsigned short* Kb  = (unsigned short*)(ws + XB + WQ + WP + QB);
  unsigned short* Vtb = (unsigned short*)(ws + XB + WQ + WP + 2*QB);
  unsigned short* Ab  = (unsigned short*)(ws + XB + WQ + WP + 3*QB);

  cvt_bf16_kernel<<<(TOKENS*DIM)/256, 256, 0, stream>>>(x, xb, TOKENS*DIM);
  cvt_bf16_kernel<<<(QKVOUT*DIM)/256, 256, 0, stream>>>(w_qkv, wqb, QKVOUT*DIM);
  cvt_bf16_kernel<<<(DIM*DIM)/256,   256, 0, stream>>>(w_proj, wpb, DIM*DIM);

  qkv_gemm_kernel<<<dim3(QKVOUT/256, TOKENS/16), 128, 0, stream>>>(xb, wqb, Qb, Kb, Vtb);
  flash_attn_kernel<<<dim3(SEQ/64, BATCH*NHEADS), 128, 0, stream>>>(Qb, Kb, Vtb, Ab);
  proj_gemm_kernel<<<dim3(DIM/256, TOKENS/16), 128, 0, stream>>>(Ab, wpb, b_proj, out);
}
